// Decoder_44959717655245
// MI455X (gfx1250) — matrix operand order repaired and checked by reading, not compiled, not run on hardware
//
#include <hip/hip_runtime.h>
#include <hip/hip_bf16.h>

// ---------------- problem dims ----------------
#define B_   64
#define P_   196
#define E_   2048
#define D_   512
#define V_   10000
#define L_   22
#define T_   21          // L-1 decode steps
#define XH_  3072        // [emb(512) | awe(2048) | h(512)]
#define G4_  2048        // 4*D LSTM gates

typedef __bf16 bf16_t;
typedef __bf16 v16bf __attribute__((ext_vector_type(16)));
typedef __bf16 v8bf  __attribute__((ext_vector_type(8)));
typedef float  v8f   __attribute__((ext_vector_type(8)));

static __device__ __forceinline__ bf16_t f2bf(float f) {
    // round-to-nearest-even fp32 -> bf16
    unsigned u = __builtin_bit_cast(unsigned, f);
    unsigned r = (u + 0x7FFFu + ((u >> 16) & 1u)) >> 16;
    unsigned short s = (unsigned short)r;
    return __builtin_bit_cast(bf16_t, s);
}
static __device__ __forceinline__ float sigmoidf_(float x) {
    return 1.0f / (1.0f + __expf(-x));
}
static __device__ __forceinline__ v16bf cat16(v8bf a, v8bf b) {
    return __builtin_shufflevector(a, b, 0, 1, 2, 3, 4, 5, 6, 7, 8, 9, 10, 11, 12, 13, 14, 15);
}
// load 16 bf16 as [p, p+8) U [p+16, p+24) (two b128 loads)
static __device__ __forceinline__ v16bf ld16bf(const bf16_t* p) {
    v8bf a = *(const v8bf*)(p);
    v8bf b = *(const v8bf*)(p + 16);
    return cat16(a, b);
}

// ---------------- elementwise / prep kernels ----------------

__global__ void k_f32_to_bf16(const float* __restrict__ src, bf16_t* __restrict__ dst, long n) {
    for (long i = blockIdx.x * (long)blockDim.x + threadIdx.x; i < n; i += (long)gridDim.x * blockDim.x)
        dst[i] = f2bf(src[i]);
}

__global__ void k_zero_bf16(bf16_t* __restrict__ dst, long n) {
    for (long i = blockIdx.x * (long)blockDim.x + threadIdx.x; i < n; i += (long)gridDim.x * blockDim.x)
        dst[i] = __builtin_bit_cast(bf16_t, (unsigned short)0);
}

// transpose+convert weight: src is (K x N) fp32 row-major, dst row n holds K bf16 at stride dstStride, offset dstOff
__global__ void k_transpose_bf16(const float* __restrict__ src, bf16_t* __restrict__ dst,
                                 int K, int N, int dstStride, int dstOff) {
    long total = (long)K * N;
    for (long i = blockIdx.x * (long)blockDim.x + threadIdx.x; i < total; i += (long)gridDim.x * blockDim.x) {
        int k = (int)(i / N);
        int n = (int)(i % N);
        dst[(long)n * dstStride + dstOff + k] = f2bf(src[i]);
    }
}

__global__ void k_bias_add(const float* __restrict__ a, const float* __restrict__ b,
                           float* __restrict__ o, int n) {
    int i = blockIdx.x * blockDim.x + threadIdx.x;
    if (i < n) o[i] = a[i] + b[i];
}

// avg over P, emitted directly as bf16 (B x E)
__global__ void k_avg_bf16(const float* __restrict__ enc, bf16_t* __restrict__ avg_bf) {
    int idx = blockIdx.x * blockDim.x + threadIdx.x;        // B*E
    if (idx >= B_ * E_) return;
    int b = idx >> 11, e = idx & (E_ - 1);
    const float* p = enc + ((long)b * P_) * E_ + e;
    float s = 0.f;
    for (int pp = 0; pp < P_; ++pp) s += p[(long)pp * E_];
    avg_bf[idx] = f2bf(s * (1.0f / (float)P_));
}

__global__ void k_h_to_bf16(const float* __restrict__ h, bf16_t* __restrict__ h_bf) {
    int i = blockIdx.x * blockDim.x + threadIdx.x;
    if (i < B_ * D_) h_bf[i] = f2bf(h[i]);
}

// copy encoded_captions and dec_len (raw int bit patterns) into flat output
__global__ void k_copy_ints(const int* __restrict__ caps, const int* __restrict__ clens,
                            int* __restrict__ out_caps, int* __restrict__ out_len) {
    int i = blockIdx.x * blockDim.x + threadIdx.x;
    if (i < B_ * L_) out_caps[i] = caps[i];
    if (i < B_)      out_len[i]  = clens[i] - 1;
}

// ---------------- WMMA bf16 GEMM ----------------
// C(MxN fp32, row stride ldc) = act( A(MxK bf16) @ W(KxN) + bias ).
// Wt is W pre-transposed (rows = N, cols = K) and MUST be allocated/filled with
// ceil(N/64)*64 rows (pad rows zeroed) so the K-loop is completely branch-free.
// Each wave owns a 16x64 tile (4 accumulators, A fragment reused 4x).
// Requires M%16==0, K%32==0, N%16==0.  act: 0=none, 1=sigmoid.
// clens!=null => zero output rows where (clens[row]-1) <= t.
__global__ __launch_bounds__(256)
void k_gemm_bf16(const bf16_t* __restrict__ A, const bf16_t* __restrict__ Wt,
                 const float* __restrict__ bias, float* __restrict__ C, long ldc,
                 int M, int N, int K, int act, const int* __restrict__ clens, int t) {
    int waveId = blockIdx.x * (blockDim.x >> 5) + (threadIdx.x >> 5);
    int lane   = threadIdx.x & 31;
    int nGroups = (N + 63) >> 6;
    int mTiles  = M >> 4;
    if (waveId >= mTiles * nGroups) return;
    int m0 = (waveId / nGroups) << 4;
    int n0 = (waveId % nGroups) << 6;
    int lm = lane & 15;
    int hi = lane >> 4;

    v8f acc0 = {}, acc1 = {}, acc2 = {}, acc3 = {};

    // A fragment: 16x32 bf16, lane (m=lm, hi) holds K in [hi*8,hi*8+8) U [16+hi*8,16+hi*8+8)
    const bf16_t* aPtr  = A + (long)(m0 + lm) * K + hi * 8;
    // B fragment: 32x16 bf16, lane (n=lm, hi) holds K in [hi*8,hi*8+8) U [16+hi*8,16+hi*8+8) of column n
    const bf16_t* bPtr0 = Wt + (long)(n0 + lm) * K + hi * 8;
    const bf16_t* bPtr1 = bPtr0 + (long)16 * K;
    const bf16_t* bPtr2 = bPtr0 + (long)32 * K;
    const bf16_t* bPtr3 = bPtr0 + (long)48 * K;

    for (int kk = 0; kk < K; kk += 32) {
        v8bf a0 = *(const v8bf*)(aPtr + kk);
        v8bf a1 = *(const v8bf*)(aPtr + kk + 16);
        v16bf af = cat16(a0, a1);
        v16bf b0 = ld16bf(bPtr0 + kk);
        v16bf b1 = ld16bf(bPtr1 + kk);
        v16bf b2 = ld16bf(bPtr2 + kk);
        v16bf b3 = ld16bf(bPtr3 + kk);
        acc0 = __builtin_amdgcn_wmma_f32_16x16x32_bf16(false, af, false, b0, (short)0, acc0, false, false);
        acc1 = __builtin_amdgcn_wmma_f32_16x16x32_bf16(false, af, false, b1, (short)0, acc1, false, false);
        acc2 = __builtin_amdgcn_wmma_f32_16x16x32_bf16(false, af, false, b2, (short)0, acc2, false, false);
        acc3 = __builtin_amdgcn_wmma_f32_16x16x32_bf16(false, af, false, b3, (short)0, acc3, false, false);
    }

    v8f accs[4] = { acc0, acc1, acc2, acc3 };
#pragma unroll
    for (int nt = 0; nt < 4; ++nt) {
        if (n0 + nt * 16 >= N) continue;          // uniform tail guard (N % 16 == 0)
        int col = n0 + nt * 16 + lm;
        float bv = bias ? bias[col] : 0.0f;
#pragma unroll
        for (int r = 0; r < 8; ++r) {
            int m = m0 + hi * 8 + r;
            float v = accs[nt][r] + bv;
            if (act == 1) v = sigmoidf_(v);
            if (clens) v = ((clens[m] - 1) > t) ? v : 0.0f;
            C[(long)m * ldc + col] = v;
        }
    }
}

// ---------------- attention / LSTM kernels ----------------

// e[b,p] = relu(enc_att[b,p,:] + dec[b,:]) . attW + att_b   — one wave per row
__global__ void k_att_score(const float* __restrict__ enc_att, const float* __restrict__ dec,
                            const float* __restrict__ attW, const float* __restrict__ att_b,
                            float* __restrict__ e) {
    int row  = blockIdx.x * (blockDim.x >> 5) + (threadIdx.x >> 5);
    if (row >= B_ * P_) return;
    int lane = threadIdx.x & 31;
    int b = row / P_;
    const float* ea = enc_att + (long)row * D_;
    const float* dv = dec + b * D_;
    float s = 0.f;
    for (int d = lane; d < D_; d += 32) {
        float x = ea[d] + dv[d];
        s += (x > 0.f ? x : 0.f) * attW[d];
    }
    for (int o = 16; o; o >>= 1) s += __shfl_xor(s, o, 32);
    if (lane == 0) e[row] = s + att_b[0];
}

// softmax over P per batch row; writes alpha (ws) and masked alphas (output)
__global__ void k_softmax_alpha(const float* __restrict__ e, float* __restrict__ alpha,
                                float* __restrict__ alphas_out, const int* __restrict__ clens, int t) {
    int b = blockIdx.x, tid = threadIdx.x;     // blockDim = 256
    __shared__ float red[256];
    float v = (tid < P_) ? e[b * P_ + tid] : -3.4e38f;
    red[tid] = v; __syncthreads();
    for (int s = 128; s; s >>= 1) { if (tid < s) red[tid] = fmaxf(red[tid], red[tid + s]); __syncthreads(); }
    float mx = red[0]; __syncthreads();
    float ex = (tid < P_) ? __expf(v - mx) : 0.f;
    red[tid] = ex; __syncthreads();
    for (int s = 128; s; s >>= 1) { if (tid < s) red[tid] += red[tid + s]; __syncthreads(); }
    float inv = 1.0f / red[0];
    if (tid < P_) {
        float a = ex * inv;
        alpha[b * P_ + tid] = a;
        bool active = (clens[b] - 1) > t;
        alphas_out[((long)b * T_ + t) * P_ + tid] = active ? a : 0.0f;
    }
}

// awe[b,e] = gate[b,e] * sum_p alpha[b,p] * enc[b,p,e]
__global__ void k_awe(const float* __restrict__ enc, const float* __restrict__ alpha,
                      const float* __restrict__ gate, float* __restrict__ awe) {
    int idx = blockIdx.x * blockDim.x + threadIdx.x;
    if (idx >= B_ * E_) return;
    int b = idx >> 11, e = idx & (E_ - 1);
    const float* ep = enc + ((long)b * P_) * E_ + e;
    const float* ap = alpha + b * P_;
    float s = 0.f;
    for (int p = 0; p < P_; ++p) s += ap[p] * ep[(long)p * E_];
    awe[idx] = gate[idx] * s;
}

// xh = [emb_t | awe | h] as bf16 (B x 3072)
__global__ void k_build_xh(const int* __restrict__ caps, const float* __restrict__ embW,
                           const float* __restrict__ awe, const float* __restrict__ h,
                           bf16_t* __restrict__ xh, int t) {
    int idx = blockIdx.x * blockDim.x + threadIdx.x;
    if (idx >= B_ * XH_) return;
    int b = idx / XH_, col = idx % XH_;
    float v;
    if (col < D_) {
        int tok = caps[b * L_ + t];
        v = embW[(long)tok * D_ + col];
    } else if (col < D_ + E_) {
        v = awe[b * E_ + (col - D_)];
    } else {
        v = h[b * D_ + (col - D_ - E_)];
    }
    xh[idx] = f2bf(v);
}

// LSTM pointwise: gates(B x 4D, blocks i|f|g|o) -> c_new/h_new; masked state update; hnew_bf for preds GEMM
__global__ void k_lstm(const float* __restrict__ gates, float* __restrict__ h, float* __restrict__ c,
                       bf16_t* __restrict__ hnew_bf, const int* __restrict__ clens, int t) {
    int idx = blockIdx.x * blockDim.x + threadIdx.x;
    if (idx >= B_ * D_) return;
    int b = idx >> 9, d = idx & (D_ - 1);
    const float* g = gates + (long)b * G4_;
    float gi = g[d], gf = g[D_ + d], gg = g[2 * D_ + d], go = g[3 * D_ + d];
    float cn = sigmoidf_(gf) * c[idx] + sigmoidf_(gi) * tanhf(gg);
    float hn = sigmoidf_(go) * tanhf(cn);
    hnew_bf[idx] = f2bf(hn);
    if ((clens[b] - 1) > t) { h[idx] = hn; c[idx] = cn; }
}

// ---------------- host launcher ----------------
extern "C" void kernel_launch(void* const* d_in, const int* in_sizes, int n_in,
                              void* d_out, int out_size, void* d_ws, size_t ws_size,
                              hipStream_t stream) {
    (void)in_sizes; (void)n_in; (void)out_size; (void)ws_size;
    const float* enc      = (const float*)d_in[0];
    const int*   caps     = (const int*)  d_in[1];
    const int*   clens    = (const int*)  d_in[2];
    const float* embW     = (const float*)d_in[3];
    const float* enc_attW = (const float*)d_in[4];
    const float* enc_attb = (const float*)d_in[5];
    const float* dec_attW = (const float*)d_in[6];
    const float* dec_attb = (const float*)d_in[7];
    const float* attW     = (const float*)d_in[8];
    const float* attb     = (const float*)d_in[9];
    const float* h_linW   = (const float*)d_in[10];
    const float* h_linb   = (const float*)d_in[11];
    const float* c_linW   = (const float*)d_in[12];
    const float* c_linb   = (const float*)d_in[13];
    const float* f_betaW  = (const float*)d_in[14];
    const float* f_betab  = (const float*)d_in[15];
    const float* W_ih     = (const float*)d_in[16];
    const float* W_hh     = (const float*)d_in[17];
    const float* b_ih     = (const float*)d_in[18];
    const float* b_hh     = (const float*)d_in[19];
    const float* fc_W     = (const float*)d_in[20];
    const float* fc_b     = (const float*)d_in[21];

    const int V_PAD = ((V_ + 63) / 64) * 64;   // 10048: fc weight rows padded so GEMM K-loop is branch-free

    // ---- workspace carve-out ----
    char* ws = (char*)d_ws;
    size_t cur = 0;
    auto alloc = [&](size_t bytes) -> char* {
        char* p = ws + cur;
        cur += (bytes + 255) & ~(size_t)255;
        return p;
    };
    bf16_t* enc_bf     = (bf16_t*)alloc((size_t)B_ * P_ * E_ * 2);
    bf16_t* enc_attWt  = (bf16_t*)alloc((size_t)D_ * E_ * 2);       // 512 rows x 2048
    bf16_t* h_linWt    = (bf16_t*)alloc((size_t)D_ * E_ * 2);
    bf16_t* c_linWt    = (bf16_t*)alloc((size_t)D_ * E_ * 2);
    bf16_t* dec_attWt  = (bf16_t*)alloc((size_t)D_ * D_ * 2);
    bf16_t* f_betaWt   = (bf16_t*)alloc((size_t)E_ * D_ * 2);       // 2048 rows x 512
    bf16_t* WcatT      = (bf16_t*)alloc((size_t)G4_ * XH_ * 2);     // 2048 rows x 3072
    bf16_t* fcWt       = (bf16_t*)alloc((size_t)V_PAD * D_ * 2);    // 10048 rows x 512 (padded)
    float*  b_comb     = (float*) alloc((size_t)G4_ * 4);
    bf16_t* avg_bf     = (bf16_t*)alloc((size_t)B_ * E_ * 2);
    float*  enc_att    = (float*) alloc((size_t)B_ * P_ * D_ * 4);
    float*  h_st       = (float*) alloc((size_t)B_ * D_ * 4);
    float*  c_st       = (float*) alloc((size_t)B_ * D_ * 4);
    bf16_t* h_bf       = (bf16_t*)alloc((size_t)B_ * D_ * 2);
    bf16_t* hnew_bf    = (bf16_t*)alloc((size_t)B_ * D_ * 2);
    float*  dec_buf    = (float*) alloc((size_t)B_ * D_ * 4);
    float*  gate_buf   = (float*) alloc((size_t)B_ * E_ * 4);
    float*  e_buf      = (float*) alloc((size_t)B_ * P_ * 4);
    float*  alpha_buf  = (float*) alloc((size_t)B_ * P_ * 4);
    float*  awe_buf    = (float*) alloc((size_t)B_ * E_ * 4);
    bf16_t* xh_buf     = (bf16_t*)alloc((size_t)B_ * XH_ * 2);
    float*  gates_buf  = (float*) alloc((size_t)B_ * G4_ * 4);

    // ---- output layout (flat, return order) ----
    float* out_f = (float*)d_out;
    const long OFF_CAPS = (long)B_ * T_ * V_;
    const long OFF_DLEN = OFF_CAPS + (long)B_ * L_;
    const long OFF_ALPH = OFF_DLEN + B_;
    float* pred_out   = out_f;
    int*   caps_out   = (int*)(out_f + OFF_CAPS);
    int*   dlen_out   = (int*)(out_f + OFF_DLEN);
    float* alphas_out = out_f + OFF_ALPH;

    auto blocks = [](long n, int bs) { return (int)((n + bs - 1) / bs); };
    auto gemm = [&](const bf16_t* A, const bf16_t* Wt, const float* bias, float* C, long ldc,
                    int M, int N, int K, int act, const int* cl, int t) {
        int waves = (M >> 4) * ((N + 63) >> 6);
        k_gemm_bf16<<<(waves + 7) / 8, 256, 0, stream>>>(A, Wt, bias, C, ldc, M, N, K, act, cl, t);
    };

    // ---- one-time (per launch) conversions ----
    k_f32_to_bf16<<<4096, 256, 0, stream>>>(enc, enc_bf, (long)B_ * P_ * E_);
    k_transpose_bf16<<<2048, 256, 0, stream>>>(enc_attW, enc_attWt, E_, D_, E_, 0);
    k_transpose_bf16<<<2048, 256, 0, stream>>>(h_linW,   h_linWt,   E_, D_, E_, 0);
    k_transpose_bf16<<<2048, 256, 0, stream>>>(c_linW,   c_linWt,   E_, D_, E_, 0);
    k_transpose_bf16<<<1024, 256, 0, stream>>>(dec_attW, dec_attWt, D_, D_, D_, 0);
    k_transpose_bf16<<<2048, 256, 0, stream>>>(f_betaW,  f_betaWt,  D_, E_, D_, 0);
    k_transpose_bf16<<<4096, 256, 0, stream>>>(W_ih, WcatT, D_ + E_, G4_, XH_, 0);
    k_transpose_bf16<<<2048, 256, 0, stream>>>(W_hh, WcatT, D_,      G4_, XH_, D_ + E_);
    k_transpose_bf16<<<4096, 256, 0, stream>>>(fc_W, fcWt, D_, V_, D_, 0);
    k_zero_bf16<<<48, 256, 0, stream>>>(fcWt + (long)V_ * D_, (long)(V_PAD - V_) * D_);
    k_bias_add<<<blocks(G4_, 256), 256, 0, stream>>>(b_ih, b_hh, b_comb, G4_);
    k_copy_ints<<<blocks(B_ * L_, 256), 256, 0, stream>>>(caps, clens, caps_out, dlen_out);

    // ---- init state: avg -> h0, c0 ; enc_att precompute ----
    k_avg_bf16<<<blocks((long)B_ * E_, 256), 256, 0, stream>>>(enc, avg_bf);
    gemm(avg_bf, h_linWt, h_linb, h_st, D_, B_, D_, E_, 0, nullptr, 0);
    gemm(avg_bf, c_linWt, c_linb, c_st, D_, B_, D_, E_, 0, nullptr, 0);
    gemm(enc_bf, enc_attWt, enc_attb, enc_att, D_, B_ * P_, D_, E_, 0, nullptr, 0);

    // ---- decode loop ----
    for (int t = 0; t < T_; ++t) {
        k_h_to_bf16<<<blocks(B_ * D_, 256), 256, 0, stream>>>(h_st, h_bf);
        gemm(h_bf, dec_attWt, dec_attb, dec_buf, D_, B_, D_, D_, 0, nullptr, 0);
        gemm(h_bf, f_betaWt, f_betab, gate_buf, E_, B_, E_, D_, 1, nullptr, 0);   // sigmoid gate
        k_att_score<<<blocks((long)B_ * P_ * 32, 256), 256, 0, stream>>>(enc_att, dec_buf, attW, attb, e_buf);
        k_softmax_alpha<<<B_, 256, 0, stream>>>(e_buf, alpha_buf, alphas_out, clens, t);
        k_awe<<<blocks((long)B_ * E_, 256), 256, 0, stream>>>(enc, alpha_buf, gate_buf, awe_buf);
        k_build_xh<<<blocks((long)B_ * XH_, 256), 256, 0, stream>>>(caps, embW, awe_buf, h_st, xh_buf, t);
        gemm(xh_buf, WcatT, b_comb, gates_buf, G4_, B_, G4_, XH_, 0, nullptr, 0);
        k_lstm<<<blocks(B_ * D_, 256), 256, 0, stream>>>(gates_buf, h_st, c_st, hnew_bf, clens, t);
        gemm(hnew_bf, fcWt, fc_b, pred_out + (long)t * V_, (long)T_ * V_, B_, V_, D_, 0, clens, t);
    }
}
